// PDG2Seq_23321672417832
// MI455X (gfx1250) — matrix operand order repaired and checked by reading, not compiled, not run on hardware
//
#include <hip/hip_runtime.h>
#include <hip/hip_bf16.h>

typedef __attribute__((ext_vector_type(16))) _Float16 v16h;
typedef __attribute__((ext_vector_type(8)))  _Float16 v8h;
typedef __attribute__((ext_vector_type(8)))  float    v8f;
typedef _Float16 f16;

#define Bc    16
#define Tc    12
#define HORc  12
#define Nn    512
#define Hh    64
#define Dd    16
#define DP    32      // D padded to 32 (K of one WMMA) — upper half zeros in memory
#define CIN   66      // IN_DIM + H = 2 + 64
#define KFLAT 132     // K * CIN
#define KP    160     // KFLAT padded to multiple of 32
#define GOUT  128     // 2*H
#define UOUT  64      // H
#define FP    80      // CIN padded to multiple of 16
#define EPSf  1e-5f

#define CDIV(a,b) (((a)+(b)-1)/(b))

// ---------------------------------------------------------------- WMMA helpers
__device__ __forceinline__ v8f wmma_f16(v16h a, v16h b, v8f c) {
  return __builtin_amdgcn_wmma_f32_16x16x32_f16(false, a, false, b, (short)0, c,
                                                false, false);
}

// A-matrix 16x32 f16 fragment: row = lane&15; halves 0..7 -> K = k0 + 8*(lane>>4)+j,
// halves 8..15 -> +16.  Two 128-bit loads (all layouts are 16B aligned).
__device__ __forceinline__ v16h load_a_frag(const f16* row, int k0, int laneHi) {
  const f16* p0 = row + k0 + laneHi * 8;
  v8h lo = *(const v8h*)(p0);
  v8h hi = *(const v8h*)(p0 + 16);
  return __builtin_shufflevector(lo, hi, 0, 1, 2, 3, 4, 5, 6, 7,
                                         8, 9, 10, 11, 12, 13, 14, 15);
}

// B-matrix 32x16 f16 fragment: col = lane&15; halves 0..7 -> K = k0 + 8*(lane>>4)+j,
// halves 8..15 -> +16.  Two 128-bit loads.
__device__ __forceinline__ v16h load_b_frag(const f16* col, int k0, int laneHi) {
  const f16* p = col + k0 + laneHi * 8;
  v8h lo = *(const v8h*)(p);
  v8h hi = *(const v8h*)(p + 16);
  return __builtin_shufflevector(lo, hi, 0, 1, 2, 3, 4, 5, 6, 7,
                                         8, 9, 10, 11, 12, 13, 14, 15);
}

// ------------------------------------------------------------ one-time prep
__global__ void k_stats(const float* __restrict__ src, const float* __restrict__ rw,
                        const float* __restrict__ rb, float* __restrict__ xn,
                        float* __restrict__ meanb, float* __restrict__ stdb) {
  int idx = blockIdx.x * blockDim.x + threadIdx.x;
  if (idx >= Bc * Nn * 2) return;
  int c = idx & 1, n = (idx >> 1) % Nn, b = idx / (Nn * 2);
  float m = 0.f, sq = 0.f;
  for (int t = 0; t < Tc; ++t) {
    float v = src[(((size_t)b * Tc + t) * Nn + n) * 4 + c];
    m += v; sq += v * v;
  }
  m /= (float)Tc;
  float var = sq / (float)Tc - m * m;
  float sd  = sqrtf(fmaxf(var, 0.f) + EPSf);
  meanb[(b * Nn + n) * 2 + c] = m;
  stdb [(b * Nn + n) * 2 + c] = sd;
  float w = rw[c], bo = rb[c];
  for (int t = 0; t < Tc; ++t) {
    float v = src[(((size_t)b * Tc + t) * Nn + n) * 4 + c];
    xn[(((size_t)b * Tc + t) * Nn + n) * 2 + c] = (v - m) / sd * w + bo;
  }
}

__global__ void k_emb(const float* __restrict__ src, const float* __restrict__ tgt,
                      const float* __restrict__ tid1, const float* __restrict__ tid2,
                      const float* __restrict__ diw1, const float* __restrict__ diw2,
                      float* __restrict__ e1e, float* __restrict__ e2e,
                      float* __restrict__ e1d, float* __restrict__ e2d) {
  int idx = blockIdx.x * blockDim.x + threadIdx.x;
  if (idx >= Bc * (Tc + HORc)) return;
  if (idx < Bc * Tc) {
    int b = idx / Tc, t = idx % Tc;
    const float* s4 = src + (((size_t)b * Tc + t) * Nn) * 4;
    int ti = (int)floorf(s4[2] * 288.f); ti = min(max(ti, 0), 287);
    int dw = (int)s4[3];                 dw = min(max(dw, 0), 6);
    for (int d = 0; d < Dd; ++d) {
      e1e[(b * Tc + t) * Dd + d] = tid1[ti * Dd + d] * diw1[dw * Dd + d];
      e2e[(b * Tc + t) * Dd + d] = tid2[ti * Dd + d] * diw2[dw * Dd + d];
    }
  } else {
    int bt = idx - Bc * Tc;
    int b = bt / HORc, t = bt % HORc;
    const float* s4 = tgt + (((size_t)b * HORc + t) * Nn) * 4;
    int ti = (int)floorf(s4[2] * 288.f); ti = min(max(ti, 0), 287);
    int dw = (int)s4[3];                 dw = min(max(dw, 0), 6);
    for (int d = 0; d < Dd; ++d) {
      e1d[(b * HORc + t) * Dd + d] = tid1[ti * Dd + d] * diw1[dw * Dd + d];
      e2d[(b * HORc + t) * Dd + d] = tid2[ti * Dd + d] * diw2[dw * Dd + d];
    }
  }
}

// W[n][o][kf] = sum_d E[n,d] * Wp[d, kf/CIN, kf%CIN, o], zero pad kf>=132, f16
__global__ void k_prepW(const float* __restrict__ E, const float* __restrict__ Wp,
                        f16* __restrict__ dst, int outDim) {
  size_t idx = (size_t)blockIdx.x * blockDim.x + threadIdx.x;
  size_t total = (size_t)Nn * outDim * KP;
  if (idx >= total) return;
  int kf = (int)(idx % KP);
  size_t t2 = idx / KP;
  int o = (int)(t2 % outDim);
  int n = (int)(t2 / outDim);
  float v = 0.f;
  if (kf < KFLAT) {
    int kk = kf / CIN, i = kf % CIN;
#pragma unroll
    for (int d = 0; d < Dd; ++d)
      v += E[n * Dd + d] * Wp[(((size_t)d * 2 + kk) * CIN + i) * outDim + o];
  }
  dst[idx] = (f16)v;
}

__global__ void k_prepB(const float* __restrict__ E, const float* __restrict__ bp,
                        float* __restrict__ dst, int outDim) {
  int idx = blockIdx.x * blockDim.x + threadIdx.x;
  if (idx >= Nn * outDim) return;
  int o = idx % outDim, n = idx / outDim;
  float v = 0.f;
#pragma unroll
  for (int d = 0; d < Dd; ++d) v += E[n * Dd + d] * bp[d * outDim + o];
  dst[idx] = v;
}

// ------------------------------------------------------------ per-step kernels
// nv1/nv2[b][n][0..15] = tanh(e*E), [16..31] = 0 (padding lives in memory so
// WMMA fragment loads are unconditional b128 loads)
__global__ void k_nv(const float* __restrict__ E, const float* __restrict__ e1a,
                     const float* __restrict__ e2a, int tIdx, int tStride,
                     f16* __restrict__ nv1, f16* __restrict__ nv2) {
  int idx = blockIdx.x * blockDim.x + threadIdx.x;
  if (idx >= Bc * Nn) return;
  int n = idx % Nn, b = idx / Nn;
  const float* e1 = e1a + ((size_t)b * tStride + tIdx) * Dd;
  const float* e2 = e2a + ((size_t)b * tStride + tIdx) * Dd;
  f16* p1 = nv1 + ((size_t)(b * Nn) + n) * DP;
  f16* p2 = nv2 + ((size_t)(b * Nn) + n) * DP;
#pragma unroll
  for (int d = 0; d < Dd; ++d) {
    float ev = E[n * Dd + d];
    p1[d] = (f16)tanhf(e1[d] * ev);
    p2[d] = (f16)tanhf(e2[d] * ev);
  }
#pragma unroll
  for (int d = Dd; d < DP; ++d) { p1[d] = (f16)0.f; p2[d] = (f16)0.f; }
}

// A[b] = softmax(relu(nv1 @ nv2^T)) — WMMA scores + LDS softmax.
// grid (32, B), block 256 (8 waves); wave w covers col-tiles 4w..4w+3.
__global__ void k_attn(const f16* __restrict__ nv1, const f16* __restrict__ nv2,
                       f16* __restrict__ A) {
  __shared__ float s[16][Nn];
  const int bb = blockIdx.y;
  const int m0 = blockIdx.x * 16;
  const int tid = threadIdx.x;
  const int w = tid >> 5, lane = tid & 31, laneHi = lane >> 4, laneLo = lane & 15;

  // A-frag from padded nv1 rows (halves 8..15 read the zero padding)
  v16h afr = load_a_frag(nv1 + ((size_t)(bb * Nn) + m0 + laneLo) * DP, 0, laneHi);
#pragma unroll
  for (int c4 = 0; c4 < 4; ++c4) {
    int ct = w * 4 + c4;
    // B-frag: halves 8..15 read the zero padding (K = 16..31)
    v16h bfr = load_b_frag(nv2 + ((size_t)(bb * Nn) + ct * 16 + laneLo) * DP,
                           0, laneHi);
    v8f c = {0.f, 0.f, 0.f, 0.f, 0.f, 0.f, 0.f, 0.f};
    c = wmma_f16(afr, bfr, c);
#pragma unroll
    for (int rr = 0; rr < 8; ++rr)
      s[rr + 8 * laneHi][ct * 16 + laneLo] = fmaxf(c[rr], 0.f);
  }
  __syncthreads();

  // softmax: two rows per wave, 16 strided elements per lane
#pragma unroll
  for (int rsel = 0; rsel < 2; ++rsel) {
    int m = w * 2 + rsel;
    float mx = 0.f;   // relu output >= 0
#pragma unroll
    for (int i = 0; i < 16; ++i) mx = fmaxf(mx, s[m][lane + 32 * i]);
    for (int off = 16; off >= 1; off >>= 1) mx = fmaxf(mx, __shfl_xor(mx, off, 32));
    float sum = 0.f;
#pragma unroll
    for (int i = 0; i < 16; ++i) sum += __expf(s[m][lane + 32 * i] - mx);
    for (int off = 16; off >= 1; off >>= 1) sum += __shfl_xor(sum, off, 32);
    float inv = 1.f / sum;
#pragma unroll
    for (int i = 0; i < 16; ++i) {
      int col = lane + 32 * i;
      A[((size_t)(bb * Nn) + m0 + m) * Nn + col] =
          (f16)(__expf(s[m][col] - mx) * inv);
    }
  }
}

// Build gate/update input: xT[b][f][n] (feature-major, FP-padded, f16) and
// P[n][b][0..65]; zero pad P[n][b][132..159].
// mode 0: feat = [x, h]; mode 1: feat = [x, z*h]
__global__ void k_buildx(const float* __restrict__ xcur, int toff, int tstride,
                         const float* __restrict__ hbuf, const float* __restrict__ zbuf,
                         int mode, f16* __restrict__ xT, f16* __restrict__ P) {
  int idx = blockIdx.x * blockDim.x + threadIdx.x;
  if (idx >= Bc * Nn) return;
  int n = idx % Nn, b = idx / Nn;
  float feat[CIN];
  const float* xp = xcur + (((size_t)b * tstride + toff) * Nn + n) * 2;
  feat[0] = xp[0]; feat[1] = xp[1];
  const float* hp = hbuf + ((size_t)(b * Nn) + n) * Hh;
  if (mode) {
    const float* zp = zbuf + ((size_t)(b * Nn) + n) * Hh;
#pragma unroll
    for (int j = 0; j < Hh; ++j) feat[2 + j] = zp[j] * hp[j];
  } else {
#pragma unroll
    for (int j = 0; j < Hh; ++j) feat[2 + j] = hp[j];
  }
#pragma unroll
  for (int f = 0; f < FP; ++f)
    xT[((size_t)b * FP + f) * Nn + n] = (f < CIN) ? (f16)feat[f] : (f16)0.f;
  f16* p = P + ((size_t)n * Bc + b) * KP;
#pragma unroll
  for (int f = 0; f < CIN; ++f) p[f] = (f16)feat[f];
#pragma unroll
  for (int f = KFLAT; f < KP; ++f) p[f] = (f16)0.f;
}

// x1 = A @ x : per b, (512x512)@(512x80). grid (32, B), block 160 (5 waves,
// one feature-tile each). Writes into P[n][b][66..131].
__global__ void k_Ax(const f16* __restrict__ A, const f16* __restrict__ xT,
                     f16* __restrict__ P) {
  const int bb = blockIdx.y, m0 = blockIdx.x * 16;
  const int tid = threadIdx.x, w = tid >> 5, lane = tid & 31;
  const int laneHi = lane >> 4, laneLo = lane & 15;
  const f16* arow = A  + ((size_t)(bb * Nn) + m0 + laneLo) * Nn;
  const f16* xcol = xT + ((size_t)(bb * FP) + w * 16 + laneLo) * Nn;
  v8f c = {0.f, 0.f, 0.f, 0.f, 0.f, 0.f, 0.f, 0.f};
#pragma unroll 4
  for (int kc = 0; kc < 16; ++kc)
    c = wmma_f16(load_a_frag(arow, kc * 32, laneHi),
                 load_b_frag(xcol, kc * 32, laneHi), c);
  int ft = w * 16 + laneLo;
  if (ft < CIN) {
#pragma unroll
    for (int rr = 0; rr < 8; ++rr) {
      int node = m0 + rr + 8 * laneHi;
      P[((size_t)node * Bc + bb) * KP + CIN + ft] = (f16)c[rr];
    }
  }
}

// Per-node gate GEMM: (16x160)@(160x128) + bias, sigmoid -> z, r.
// grid 512 (node), block 256 (8 out-tiles of 16).
__global__ void k_gate(const f16* __restrict__ P, const f16* __restrict__ W,
                       const float* __restrict__ bias,
                       float* __restrict__ zbuf, float* __restrict__ rbuf) {
  const int n = blockIdx.x;
  const int tid = threadIdx.x, w = tid >> 5, lane = tid & 31;
  const int laneHi = lane >> 4, laneLo = lane & 15;
  const int o0 = w * 16;
  const f16* arow = P + ((size_t)n * Bc + laneLo) * KP;
  const f16* bcol = W + ((size_t)n * GOUT + o0 + laneLo) * KP;
  if (n + 1 < Nn)
    __builtin_prefetch(W + ((size_t)(n + 1) * GOUT + o0 + laneLo) * KP, 0, 1);
  v8f c = {0.f, 0.f, 0.f, 0.f, 0.f, 0.f, 0.f, 0.f};
#pragma unroll
  for (int kc = 0; kc < 5; ++kc)
    c = wmma_f16(load_a_frag(arow, kc * 32, laneHi),
                 load_b_frag(bcol, kc * 32, laneHi), c);
  int o = o0 + laneLo;
#pragma unroll
  for (int rr = 0; rr < 8; ++rr) {
    int b = rr + 8 * laneHi;
    float v = c[rr] + bias[n * GOUT + o];
    v = 1.f / (1.f + __expf(-v));
    if (o < Hh) zbuf[((size_t)(b * Nn) + n) * Hh + o] = v;
    else        rbuf[((size_t)(b * Nn) + n) * Hh + (o - Hh)] = v;
  }
}

// Per-node update GEMM: (16x160)@(160x64) + bias, tanh -> hc; h = r*h + (1-r)*hc.
// grid 512, block 128 (4 out-tiles).
__global__ void k_upd(const f16* __restrict__ P, const f16* __restrict__ W,
                      const float* __restrict__ bias, const float* __restrict__ rbuf,
                      float* __restrict__ hbuf) {
  const int n = blockIdx.x;
  const int tid = threadIdx.x, w = tid >> 5, lane = tid & 31;
  const int laneHi = lane >> 4, laneLo = lane & 15;
  const int o0 = w * 16;
  const f16* arow = P + ((size_t)n * Bc + laneLo) * KP;
  const f16* bcol = W + ((size_t)n * UOUT + o0 + laneLo) * KP;
  if (n + 1 < Nn)
    __builtin_prefetch(W + ((size_t)(n + 1) * UOUT + o0 + laneLo) * KP, 0, 1);
  v8f c = {0.f, 0.f, 0.f, 0.f, 0.f, 0.f, 0.f, 0.f};
#pragma unroll
  for (int kc = 0; kc < 5; ++kc)
    c = wmma_f16(load_a_frag(arow, kc * 32, laneHi),
                 load_b_frag(bcol, kc * 32, laneHi), c);
  int o = o0 + laneLo;
#pragma unroll
  for (int rr = 0; rr < 8; ++rr) {
    int b = rr + 8 * laneHi;
    float hc = tanhf(c[rr] + bias[n * UOUT + o]);
    size_t hi = ((size_t)(b * Nn) + n) * Hh + o;
    float rv = rbuf[hi];
    float ho = hbuf[hi];
    hbuf[hi] = rv * ho + (1.f - rv) * hc;
  }
}

// go = h @ projW + projb; write denormalized output row td
__global__ void k_proj(const float* __restrict__ hbuf, const float* __restrict__ pW,
                       const float* __restrict__ pb, const float* __restrict__ rw,
                       const float* __restrict__ rb, const float* __restrict__ meanb,
                       const float* __restrict__ stdb, float* __restrict__ gob,
                       float* __restrict__ outp, int td) {
  int idx = blockIdx.x * blockDim.x + threadIdx.x;
  if (idx >= Bc * Nn) return;
  int n = idx % Nn, b = idx / Nn;
  const float* hp = hbuf + ((size_t)(b * Nn) + n) * Hh;
#pragma unroll
  for (int o = 0; o < 2; ++o) {
    float g = pb[o];
#pragma unroll
    for (int j = 0; j < Hh; ++j) g += hp[j] * pW[j * 2 + o];
    gob[((size_t)(b * Nn) + n) * 2 + o] = g;
    float sd = stdb [(b * Nn + n) * 2 + o];
    float mn = meanb[(b * Nn + n) * 2 + o];
    outp[(((size_t)b * HORc + td) * Nn + n) * 2 + o] =
        (g - rb[o]) / (rw[o] + EPSf * EPSf) * sd + mn;
  }
}

// ------------------------------------------------------------------ launcher
extern "C" void kernel_launch(void* const* d_in, const int* in_sizes, int n_in,
                              void* d_out, int out_size, void* d_ws, size_t ws_size,
                              hipStream_t stream) {
  (void)in_sizes; (void)n_in; (void)out_size; (void)ws_size;
  const float* src  = (const float*)d_in[0];
  const float* tgt  = (const float*)d_in[1];
  const float* E    = (const float*)d_in[2];
  const float* tid1 = (const float*)d_in[3];
  const float* tid2 = (const float*)d_in[4];
  const float* diw1 = (const float*)d_in[5];
  const float* diw2 = (const float*)d_in[6];
  const float* rw   = (const float*)d_in[7];
  const float* rb   = (const float*)d_in[8];
  const float* egW  = (const float*)d_in[9];
  const float* egb  = (const float*)d_in[10];
  const float* euW  = (const float*)d_in[11];
  const float* eub  = (const float*)d_in[12];
  const float* dgW  = (const float*)d_in[13];
  const float* dgb  = (const float*)d_in[14];
  const float* duW  = (const float*)d_in[15];
  const float* dub  = (const float*)d_in[16];
  const float* pW   = (const float*)d_in[17];
  const float* pb   = (const float*)d_in[18];
  float* outp = (float*)d_out;

  char* base = (char*)d_ws; size_t off = 0;
  auto alloc = [&](size_t bytes) -> char* {
    char* p = base + off; off = (off + bytes + 255) & ~(size_t)255; return p;
  };
  f16*   WgE  = (f16*)  alloc((size_t)Nn * GOUT * KP * 2);
  f16*   WuE  = (f16*)  alloc((size_t)Nn * UOUT * KP * 2);
  f16*   WgD  = (f16*)  alloc((size_t)Nn * GOUT * KP * 2);
  f16*   WuD  = (f16*)  alloc((size_t)Nn * UOUT * KP * 2);
  float* bgE  = (float*)alloc((size_t)Nn * GOUT * 4);
  float* buE  = (float*)alloc((size_t)Nn * UOUT * 4);
  float* bgD  = (float*)alloc((size_t)Nn * GOUT * 4);
  float* buD  = (float*)alloc((size_t)Nn * UOUT * 4);
  f16*   Abuf = (f16*)  alloc((size_t)Bc * Nn * Nn * 2);
  f16*   nv1  = (f16*)  alloc((size_t)Bc * Nn * DP * 2);
  f16*   nv2  = (f16*)  alloc((size_t)Bc * Nn * DP * 2);
  float* hbuf = (float*)alloc((size_t)Bc * Nn * Hh * 4);
  float* zbuf = (float*)alloc((size_t)Bc * Nn * Hh * 4);
  float* rbuf = (float*)alloc((size_t)Bc * Nn * Hh * 4);
  float* xn   = (float*)alloc((size_t)Bc * Tc * Nn * 2 * 4);
  float* mnb  = (float*)alloc((size_t)Bc * Nn * 2 * 4);
  float* sdb  = (float*)alloc((size_t)Bc * Nn * 2 * 4);
  float* e1e  = (float*)alloc((size_t)Bc * Tc * Dd * 4);
  float* e2e  = (float*)alloc((size_t)Bc * Tc * Dd * 4);
  float* e1d  = (float*)alloc((size_t)Bc * HORc * Dd * 4);
  float* e2d  = (float*)alloc((size_t)Bc * HORc * Dd * 4);
  f16*   xsT  = (f16*)  alloc((size_t)Bc * FP * Nn * 2);
  f16*   xuT  = (f16*)  alloc((size_t)Bc * FP * Nn * 2);
  f16*   PG   = (f16*)  alloc((size_t)Nn * Bc * KP * 2);
  f16*   PU   = (f16*)  alloc((size_t)Nn * Bc * KP * 2);
  float* gob  = (float*)alloc((size_t)Bc * Nn * 2 * 4);

  hipMemsetAsync(hbuf, 0, (size_t)Bc * Nn * Hh * 4, stream);
  hipMemsetAsync(gob,  0, (size_t)Bc * Nn * 2 * 4, stream);

  // one-time prep
  k_stats<<<CDIV(Bc * Nn * 2, 256), 256, 0, stream>>>(src, rw, rb, xn, mnb, sdb);
  k_emb<<<CDIV(Bc * (Tc + HORc), 256), 256, 0, stream>>>(src, tgt, tid1, tid2,
                                                         diw1, diw2, e1e, e2e, e1d, e2d);
  k_prepW<<<CDIV((size_t)Nn * GOUT * KP, 256), 256, 0, stream>>>(E, egW, WgE, GOUT);
  k_prepW<<<CDIV((size_t)Nn * UOUT * KP, 256), 256, 0, stream>>>(E, euW, WuE, UOUT);
  k_prepW<<<CDIV((size_t)Nn * GOUT * KP, 256), 256, 0, stream>>>(E, dgW, WgD, GOUT);
  k_prepW<<<CDIV((size_t)Nn * UOUT * KP, 256), 256, 0, stream>>>(E, duW, WuD, UOUT);
  k_prepB<<<CDIV(Nn * GOUT, 256), 256, 0, stream>>>(E, egb, bgE, GOUT);
  k_prepB<<<CDIV(Nn * UOUT, 256), 256, 0, stream>>>(E, eub, buE, UOUT);
  k_prepB<<<CDIV(Nn * GOUT, 256), 256, 0, stream>>>(E, dgb, bgD, GOUT);
  k_prepB<<<CDIV(Nn * UOUT, 256), 256, 0, stream>>>(E, dub, buD, UOUT);

  const dim3 gTile(32, Bc);

  // encoder scan
  for (int t = 0; t < Tc; ++t) {
    k_nv<<<CDIV(Bc * Nn, 256), 256, 0, stream>>>(E, e1e, e2e, t, Tc, nv1, nv2);
    k_attn<<<gTile, 256, 0, stream>>>(nv1, nv2, Abuf);
    k_buildx<<<CDIV(Bc * Nn, 256), 256, 0, stream>>>(xn, t, Tc, hbuf, zbuf, 0, xsT, PG);
    k_Ax<<<gTile, 160, 0, stream>>>(Abuf, xsT, PG);
    k_gate<<<Nn, 256, 0, stream>>>(PG, WgE, bgE, zbuf, rbuf);
    k_buildx<<<CDIV(Bc * Nn, 256), 256, 0, stream>>>(xn, t, Tc, hbuf, zbuf, 1, xuT, PU);
    k_Ax<<<gTile, 160, 0, stream>>>(Abuf, xuT, PU);
    k_upd<<<Nn, 128, 0, stream>>>(PU, WuE, buE, rbuf, hbuf);
  }

  // decoder scan
  for (int t = 0; t < HORc; ++t) {
    k_nv<<<CDIV(Bc * Nn, 256), 256, 0, stream>>>(E, e1d, e2d, t, HORc, nv1, nv2);
    k_attn<<<gTile, 256, 0, stream>>>(nv1, nv2, Abuf);
    k_buildx<<<CDIV(Bc * Nn, 256), 256, 0, stream>>>(gob, 0, 1, hbuf, zbuf, 0, xsT, PG);
    k_Ax<<<gTile, 160, 0, stream>>>(Abuf, xsT, PG);
    k_gate<<<Nn, 256, 0, stream>>>(PG, WgD, bgD, zbuf, rbuf);
    k_buildx<<<CDIV(Bc * Nn, 256), 256, 0, stream>>>(gob, 0, 1, hbuf, zbuf, 1, xuT, PU);
    k_Ax<<<gTile, 160, 0, stream>>>(Abuf, xuT, PU);
    k_upd<<<Nn, 128, 0, stream>>>(PU, WuD, buD, rbuf, hbuf);
    k_proj<<<CDIV(Bc * Nn, 256), 256, 0, stream>>>(hbuf, pW, pb, rw, rb, mnb, sdb,
                                                   gob, outp, t);
  }
}
